// LowDimProjectedAttention_5342939316810
// MI455X (gfx1250) — compile-verified, matrix operand order checked by reading, not run on hardware
//
#include <hip/hip_runtime.h>
#include <hip/hip_bf16.h>

#define D_MODEL 2048
#define R_DIM   512
#define NHEADS  16
#define DK      32
#define BATCH   2
#define SEQ     2048
#define NTOK    (BATCH * SEQ)
#define KVTILE  1024          // 32 keys x 32 dims, bf16 elements per K/V tile

typedef __attribute__((ext_vector_type(16))) __bf16        v16bf;
typedef __attribute__((ext_vector_type(8)))  float         v8f;
typedef __attribute__((ext_vector_type(4)))  unsigned int  v4u;
typedef __attribute__((ext_vector_type(8)))  int           v8i;
typedef __attribute__((ext_vector_type(4)))  int           v4i;

#if defined(__has_builtin)
#if __has_builtin(__builtin_amdgcn_tensor_load_to_lds) && \
    __has_builtin(__builtin_amdgcn_s_wait_tensorcnt)
#define USE_TDM 1
#endif
#endif

union FragBF {
  v16bf v;
  uint4 q[2];
  unsigned short us[16];
};

__device__ __forceinline__ unsigned short f2bf(float f) {
  union { float f; unsigned u; } a;
  a.f = f;
  unsigned u = a.u;
  u += 0x7fffu + ((u >> 16) & 1u);   // round-to-nearest-even
  return (unsigned short)(u >> 16);
}

__device__ __forceinline__ unsigned pack2bf(float lo, float hi) {
  return ((unsigned)f2bf(hi) << 16) | (unsigned)f2bf(lo);
}

__device__ __forceinline__ v8f wmma_bf16(const FragBF& a, const FragBF& b, v8f c) {
  return __builtin_amdgcn_wmma_f32_16x16x32_bf16(false, a.v, false, b.v,
                                                 (short)0, c, false, false);
}

#if USE_TDM
// Issue a TDM copy of `nelem` contiguous bf16 elements global -> LDS.
// D# per CDNA5 ISA ch.8: group0 = {count/type, lds_addr, global_addr}, group1 =
// {data_size=2B, tensor_dim0 = tile_dim0 = nelem (1-D tile), stride0 = nelem},
// groups 2/3 unused (tensor <= 2-D). 6-arg builtin (this toolchain): the fifth
// int32x8 operand is the extra descriptor group, zero for 1-D tiles.
__device__ __forceinline__ void tdm_issue_tile(const unsigned short* gsrc,
                                               unsigned short* ldst, int nelem) {
  const unsigned long long ga = (unsigned long long)(size_t)gsrc;
  const unsigned lds_addr = (unsigned)(size_t)ldst;   // flat addr[31:0] == LDS addr
  v4u g0;
  g0[0] = 1u;                                          // count=1, user descriptor
  g0[1] = lds_addr;                                    // bits [63:32]
  g0[2] = (unsigned)(ga & 0xffffffffu);                // global_addr[31:0]
  g0[3] = (unsigned)((ga >> 32) & 0x01ffffffu)         // global_addr[56:32]
        | 0x80000000u;                                 // type=2 (bits 127:126)
  v8i g1;
  g1[0] = (int)(1u << 16);                             // data_size=1 -> 2 bytes
  g1[1] = (int)(((unsigned)nelem & 0xffffu) << 16);    // tensor_dim0[15:0]
  g1[2] = (int)((unsigned)nelem >> 16);                // tensor_dim0[31:16]
  g1[3] = (int)(((unsigned)nelem & 0xffffu) << 16);    // tile_dim0
  g1[4] = 0;                                           // tile_dim1/2 = 0 (unused)
  g1[5] = nelem;                                       // tensor_dim0_stride[31:0]
  g1[6] = 0;
  g1[7] = 0;
  v4i gz4 = {0, 0, 0, 0};
  v8i gz8 = {0, 0, 0, 0, 0, 0, 0, 0};
  __builtin_amdgcn_tensor_load_to_lds(g0, g1, gz4, gz4, gz8, 0);
}
#endif

// ---------------------------------------------------------------- fp32 -> bf16
__global__ void cvt_bf16_kernel(const float* __restrict__ src,
                                unsigned short* __restrict__ dst, int n) {
  int i = blockIdx.x * blockDim.x + threadIdx.x;
  if (i < n) dst[i] = f2bf(src[i]);
}

// ------------------------------------------------------- fused QKV projection
__global__ __launch_bounds__(256)
void proj_qkv_kernel(const float* __restrict__ x,
                     const unsigned short* __restrict__ Wqb,
                     const unsigned short* __restrict__ Wkb,
                     const unsigned short* __restrict__ Wvb,
                     const float* __restrict__ bq,
                     const float* __restrict__ bk,
                     const float* __restrict__ bv,
                     unsigned short* __restrict__ Qb,
                     unsigned short* __restrict__ Kb,
                     unsigned short* __restrict__ Vb) {
  __shared__ __align__(16) unsigned short lds_x[16][32];
  const int tid  = threadIdx.x;
  const int wave = tid >> 5;
  const int lane = tid & 31;
  const int mrow = lane & 15;
  const int kh   = lane >> 4;
  const int m0   = blockIdx.x * 16;
  const int which = blockIdx.y >> 1;
  const int nbase = (blockIdx.y & 1) * 256 + wave * 32;

  const unsigned short* W   = (which == 0) ? Wqb : ((which == 1) ? Wkb : Wvb);
  const float*          bia = (which == 0) ? bq  : ((which == 1) ? bk  : bv);
  unsigned short*       Out = (which == 0) ? Qb  : ((which == 1) ? Kb  : Vb);

  v8f acc0 = {}; v8f acc1 = {};
  const int r  = tid >> 4;
  const int c2 = (tid & 15) * 2;

  for (int k0 = 0; k0 < D_MODEL; k0 += 32) {
    float2 f2 = *(const float2*)(x + (size_t)(m0 + r) * D_MODEL + k0 + c2);
    *(unsigned*)&lds_x[r][c2] = pack2bf(f2.x, f2.y);
    __syncthreads();

    FragBF a, b0, b1;
    a.q[0] = *(const uint4*)&lds_x[mrow][kh * 8];
    a.q[1] = *(const uint4*)&lds_x[mrow][16 + kh * 8];
    const unsigned short* wp0 = W + (size_t)(nbase + mrow) * D_MODEL + k0;
    const unsigned short* wp1 = W + (size_t)(nbase + 16 + mrow) * D_MODEL + k0;
    b0.q[0] = *(const uint4*)(wp0 + kh * 8);
    b0.q[1] = *(const uint4*)(wp0 + 16 + kh * 8);
    b1.q[0] = *(const uint4*)(wp1 + kh * 8);
    b1.q[1] = *(const uint4*)(wp1 + 16 + kh * 8);
    __builtin_prefetch(wp0 + 32, 0, 1);
    __builtin_prefetch(wp1 + 32, 0, 1);

    acc0 = wmma_bf16(a, b0, acc0);
    acc1 = wmma_bf16(a, b1, acc1);
    __syncthreads();
  }

#pragma unroll
  for (int v = 0; v < 8; ++v) {
    const int t  = m0 + v + 8 * kh;
    const int bb = t >> 11;
    const int ss = t & (SEQ - 1);
#pragma unroll
    for (int g = 0; g < 2; ++g) {
      const int col = nbase + g * 16 + mrow;
      const float val = (g ? acc1[v] : acc0[v]) + bia[col];
      const int hh = col >> 5, dd = col & 31;
      Out[(((size_t)bb * NHEADS + hh) * SEQ + ss) * DK + dd] = f2bf(val);
    }
  }
}

// ------------------------------------------------------------- flash attention
// grid (SEQ/128, B*H); 8 waves; K/V tiles double-buffered in LDS, staged by the
// Tensor Data Mover when available (TENSORcnt), else cooperative vector loads.
__global__ __launch_bounds__(256)
void attn_kernel(const unsigned short* __restrict__ Qb,
                 const unsigned short* __restrict__ Kb,
                 const unsigned short* __restrict__ Vb,
                 const int* __restrict__ mask,
                 unsigned short* __restrict__ attnO) {
  __shared__ __align__(16) unsigned short kv[2][2][32][32]; // [buf][K|V][key][d]
  __shared__ __align__(16) unsigned short pbuf[8][16][32];  // per-wave P tile

  const int tid  = threadIdx.x;
  const int wave = tid >> 5;
  const int lane = tid & 31;
  const int mrow = lane & 15;
  const int kh   = lane >> 4;

  const int bh = blockIdx.y;
  const int b  = bh >> 4;
  const int h  = bh & 15;
  const int qbase = blockIdx.x * 128 + wave * 16;

  const unsigned short* Qp = Qb + (size_t)bh * SEQ * DK;
  const unsigned short* Kp = Kb + (size_t)bh * SEQ * DK;
  const unsigned short* Vp = Vb + (size_t)bh * SEQ * DK;

  FragBF qa;                                  // Q A-fragment, reused all steps
  {
    const unsigned short* qrow = Qp + (size_t)(qbase + mrow) * DK;
    qa.q[0] = *(const uint4*)(qrow + kh * 8);
    qa.q[1] = *(const uint4*)(qrow + 16 + kh * 8);
  }

  v8f acc0 = {}; v8f acc1 = {};
  float mi[8], li[8];
#pragma unroll
  for (int v = 0; v < 8; ++v) { mi[v] = -__builtin_inff(); li[v] = 0.0f; }

  const float scale = 0.1767766952966369f;    // 1/sqrt(DK)
  const int nsteps = SEQ / 32;

#if USE_TDM
  if (wave == 0) {                            // prologue: stage tile 0
    tdm_issue_tile(Kp, &kv[0][0][0][0], KVTILE);
    tdm_issue_tile(Vp, &kv[0][1][0][0], KVTILE);
  }
#endif

  for (int it = 0; it < nsteps; ++it) {
    const int cur = it & 1;
    const int kk  = it * 32;

#if USE_TDM
    __syncthreads();                          // all waves done with buf[1-cur]
    if (wave == 0) {
      if (it + 1 < nsteps) {
        tdm_issue_tile(Kp + (size_t)(kk + 32) * DK, &kv[1 - cur][0][0][0], KVTILE);
        tdm_issue_tile(Vp + (size_t)(kk + 32) * DK, &kv[1 - cur][1][0][0], KVTILE);
        __builtin_amdgcn_s_wait_tensorcnt(2); // tile `it` landed, next in flight
      } else {
        __builtin_amdgcn_s_wait_tensorcnt(0);
      }
    }
    __syncthreads();                          // publish tile `it`
#else
    {
      const int key = tid >> 3;
      const int d4  = (tid & 7) * 4;
      *(uint2*)&kv[cur][0][key][d4] = *(const uint2*)(Kp + (size_t)(kk + key) * DK + d4);
      *(uint2*)&kv[cur][1][key][d4] = *(const uint2*)(Vp + (size_t)(kk + key) * DK + d4);
    }
    __syncthreads();                          // double-buffered: one barrier/step
#endif

    const unsigned short (*Kt)[32] = kv[cur][0];
    const unsigned short (*Vt)[32] = kv[cur][1];

    // scores: S(16x32) = Q(16x32) x K^T, two 16x16x32 WMMAs
    FragBF kb0, kb1;
    kb0.q[0] = *(const uint4*)&Kt[mrow][kh * 8];
    kb0.q[1] = *(const uint4*)&Kt[mrow][16 + kh * 8];
    kb1.q[0] = *(const uint4*)&Kt[16 + mrow][kh * 8];
    kb1.q[1] = *(const uint4*)&Kt[16 + mrow][16 + kh * 8];
    v8f z = {};
    v8f s0 = wmma_bf16(qa, kb0, z);
    v8f s1 = wmma_bf16(qa, kb1, z);

    const float pen0 = mask[b * SEQ + kk + mrow]      ? 0.0f : -__builtin_inff();
    const float pen1 = mask[b * SEQ + kk + 16 + mrow] ? 0.0f : -__builtin_inff();

    // online softmax; rows live in 16-lane halves (row = v + 8*kh, col = lane&15)
#pragma unroll
    for (int v = 0; v < 8; ++v) {
      float fs0 = s0[v] * scale + pen0;
      float fs1 = s1[v] * scale + pen1;
      float rm = fmaxf(fs0, fs1);
#pragma unroll
      for (int off = 1; off < 16; off <<= 1)
        rm = fmaxf(rm, __shfl_xor(rm, off, 16));
      const float mnew  = fmaxf(mi[v], rm);
      const float alpha = (mi[v] == -__builtin_inff()) ? 0.0f : __expf(mi[v] - mnew);
      float e0 = (mnew == -__builtin_inff()) ? 0.0f : __expf(fs0 - mnew);
      float e1 = (mnew == -__builtin_inff()) ? 0.0f : __expf(fs1 - mnew);
      float rs = e0 + e1;
#pragma unroll
      for (int off = 1; off < 16; off <<= 1)
        rs += __shfl_xor(rs, off, 16);
      li[v] = li[v] * alpha + rs;
      mi[v] = mnew;
      acc0[v] *= alpha;
      acc1[v] *= alpha;
      pbuf[wave][v + 8 * kh][mrow]      = f2bf(e0);
      pbuf[wave][v + 8 * kh][16 + mrow] = f2bf(e1);
    }

    // O += P(16x32) x V(32x32); V consumed row-major (per-element gather)
    FragBF pa, vb0, vb1;
    pa.q[0] = *(const uint4*)&pbuf[wave][mrow][kh * 8];
    pa.q[1] = *(const uint4*)&pbuf[wave][mrow][16 + kh * 8];
#pragma unroll
    for (int e = 0; e < 8; ++e) {
      const int key0 = kh * 8 + e;
      const int key1 = 16 + kh * 8 + e;
      vb0.us[e]     = Vt[key0][mrow];
      vb0.us[8 + e] = Vt[key1][mrow];
      vb1.us[e]     = Vt[key0][16 + mrow];
      vb1.us[8 + e] = Vt[key1][16 + mrow];
    }
    acc0 = wmma_bf16(pa, vb0, acc0);
    acc1 = wmma_bf16(pa, vb1, acc1);
  }

#pragma unroll
  for (int v = 0; v < 8; ++v) {
    const float inv = li[v] > 0.0f ? 1.0f / li[v] : 0.0f;
    const int s = qbase + v + 8 * kh;
    const size_t rowoff = ((size_t)(b * SEQ + s)) * R_DIM + h * DK;
    attnO[rowoff + mrow]      = f2bf(acc0[v] * inv);
    attnO[rowoff + 16 + mrow] = f2bf(acc1[v] * inv);
  }
}

// ----------------------------------------------------------- output projection
__global__ __launch_bounds__(256)
void out_proj_kernel(const unsigned short* __restrict__ A,
                     const unsigned short* __restrict__ Wob,
                     const float* __restrict__ bo,
                     float* __restrict__ out) {
  __shared__ __align__(16) unsigned short lds_a[16][32];
  const int tid  = threadIdx.x;
  const int wave = tid >> 5;
  const int lane = tid & 31;
  const int mrow = lane & 15;
  const int kh   = lane >> 4;
  const int m0   = blockIdx.x * 16;
  const int nbase = blockIdx.y * 256 + wave * 32;

  v8f acc0 = {}; v8f acc1 = {};
  const int r  = tid >> 4;
  const int c2 = (tid & 15) * 2;

  for (int k0 = 0; k0 < R_DIM; k0 += 32) {
    unsigned av = *(const unsigned*)(A + (size_t)(m0 + r) * R_DIM + k0 + c2);
    *(unsigned*)&lds_a[r][c2] = av;
    __syncthreads();

    FragBF a, b0, b1;
    a.q[0] = *(const uint4*)&lds_a[mrow][kh * 8];
    a.q[1] = *(const uint4*)&lds_a[mrow][16 + kh * 8];
    const unsigned short* wp0 = Wob + (size_t)(nbase + mrow) * R_DIM + k0;
    const unsigned short* wp1 = Wob + (size_t)(nbase + 16 + mrow) * R_DIM + k0;
    b0.q[0] = *(const uint4*)(wp0 + kh * 8);
    b0.q[1] = *(const uint4*)(wp0 + 16 + kh * 8);
    b1.q[0] = *(const uint4*)(wp1 + kh * 8);
    b1.q[1] = *(const uint4*)(wp1 + 16 + kh * 8);
    __builtin_prefetch(wp0 + 32, 0, 1);
    __builtin_prefetch(wp1 + 32, 0, 1);

    acc0 = wmma_bf16(a, b0, acc0);
    acc1 = wmma_bf16(a, b1, acc1);
    __syncthreads();
  }

#pragma unroll
  for (int v = 0; v < 8; ++v) {
    const int t = m0 + v + 8 * kh;
#pragma unroll
    for (int g = 0; g < 2; ++g) {
      const int col = nbase + g * 16 + mrow;
      out[(size_t)t * D_MODEL + col] = (g ? acc1[v] : acc0[v]) + bo[col];
    }
  }
}

// ---------------------------------------------------------------------- launch
extern "C" void kernel_launch(void* const* d_in, const int* in_sizes, int n_in,
                              void* d_out, int out_size, void* d_ws, size_t ws_size,
                              hipStream_t stream) {
  const float* x  = (const float*)d_in[0];
  const float* Wq = (const float*)d_in[1];
  const float* bq = (const float*)d_in[2];
  const float* Wk = (const float*)d_in[3];
  const float* bk = (const float*)d_in[4];
  const float* Wv = (const float*)d_in[5];
  const float* bv = (const float*)d_in[6];
  const float* Wo = (const float*)d_in[7];
  const float* bo = (const float*)d_in[8];
  const int* mask = (const int*)d_in[9];
  float* out = (float*)d_out;

  char* ws = (char*)d_ws;
  unsigned short* Qb  = (unsigned short*)(ws + 0);           // 4 MiB (B,H,S,DK)
  unsigned short* Kb  = (unsigned short*)(ws + (4u << 20));
  unsigned short* Vb  = (unsigned short*)(ws + (8u << 20));
  unsigned short* Ab  = (unsigned short*)(ws + (12u << 20)); // (NTOK,R) bf16
  unsigned short* Wqb = (unsigned short*)(ws + (16u << 20)); // 2 MiB each
  unsigned short* Wkb = (unsigned short*)(ws + (18u << 20));
  unsigned short* Wvb = (unsigned short*)(ws + (20u << 20));
  unsigned short* Wob = (unsigned short*)(ws + (22u << 20));

  const int nW = R_DIM * D_MODEL;
  cvt_bf16_kernel<<<nW / 256, 256, 0, stream>>>(Wq, Wqb, nW);
  cvt_bf16_kernel<<<nW / 256, 256, 0, stream>>>(Wk, Wkb, nW);
  cvt_bf16_kernel<<<nW / 256, 256, 0, stream>>>(Wv, Wvb, nW);
  cvt_bf16_kernel<<<nW / 256, 256, 0, stream>>>(Wo, Wob, nW);

  proj_qkv_kernel<<<dim3(NTOK / 16, 6), 256, 0, stream>>>(
      x, Wqb, Wkb, Wvb, bq, bk, bv, Qb, Kb, Vb);

  attn_kernel<<<dim3(SEQ / 128, BATCH * NHEADS), 256, 0, stream>>>(
      Qb, Kb, Vb, mask, Ab);

  out_proj_kernel<<<dim3(NTOK / 16, D_MODEL / 256), 256, 0, stream>>>(
      Ab, Wob, bo, out);
}
